// DeformableTransformerDecoderLayer_60516089200685
// MI455X (gfx1250) — compile-verified, matrix operand order checked by reading, not run on hardware
//
#include <hip/hip_runtime.h>
#include <hip/hip_bf16.h>

typedef __attribute__((ext_vector_type(16))) _Float16 v16h;
typedef __attribute__((ext_vector_type(8)))  _Float16 v8h;
typedef __attribute__((ext_vector_type(8)))  float    v8f;
typedef __attribute__((ext_vector_type(4)))  unsigned int v4u;
typedef __attribute__((ext_vector_type(8)))  int      v8i;
typedef __attribute__((ext_vector_type(4)))  int      v4i;

#define D_MODEL 256
#define NHEAD   8
#define NLVL    4
#define NPT     4
#define DHEAD   32
#define DFF     1024
#define BATCH   8
#define LQRY    900
#define LINV    13294
#define LQP     928   /* 29 * 32, padded key length */
#define NKT     29    /* key tiles of 32 */
#define NTOK    (BATCH * LQRY)          /* 7200  */
#define NVAL    (BATCH * LINV)          /* 106352 */

#if defined(__has_builtin)
#  if __has_builtin(__builtin_amdgcn_tensor_load_to_lds)
#    define HAVE_TDM 1
#  endif
#endif
#ifndef HAVE_TDM
#  define HAVE_TDM 0
#endif

// ---------------------------------------------------------------------------
// Elementwise: out_f16 = f16(a + b)   (b may be null -> pure convert)
// ---------------------------------------------------------------------------
__global__ void addcvt_kernel(const float* __restrict__ a,
                              const float* __restrict__ b,
                              _Float16* __restrict__ out, int n) {
    int i = blockIdx.x * blockDim.x + threadIdx.x;
    if (i < n) {
        float v = a[i];
        if (b) v += b[i];
        out[i] = (_Float16)v;
    }
}

#if HAVE_TDM
// ---------------------------------------------------------------------------
// TDM: async DMA of a 64-row x 64-byte f16 tile (row stride K elements) into
// LDS with hardware padding of 4 DWORDs after every 16 DWORDs, reproducing
// the [64][40] half layout. Issued once per wave (EXEC ignored by TDM).
// Descriptor per CDNA5 ISA ch.8: group0 {count,lds,global,type}, group1
// {data_size=2B, pad_enable, pad_interval=3(16dw), pad_amount=3(4dw), dims}.
// OOB rows (tensor_dim1 = rows_rem) are zero-filled by the TDM.
// ---------------------------------------------------------------------------
__device__ __forceinline__ void tdm_load_a_tile(const _Float16* gsrc,
                                                void* ldst,
                                                unsigned rows_rem,
                                                unsigned k_rem,
                                                unsigned K) {
    unsigned lds = (unsigned)(uintptr_t)ldst;
    unsigned long long ga = (unsigned long long)(uintptr_t)gsrc;
    v4u g0 = { 1u,                                   // count = 1 valid descriptor
               lds,                                  // lds_addr
               (unsigned)ga,                         // global_addr[31:0]
               (unsigned)((ga >> 32) & 0x1FFFFFFu) | (2u << 30) };  // [56:32]|type=2
    const unsigned w0 = (1u << 16)      // data_size = 2 bytes
                      | (1u << 20)      // pad_enable
                      | (3u << 22)      // pad_interval: 16 DWORDs
                      | (3u << 25);     // pad_amount : 4 DWORDs
    v8i g1 = { (int)w0,
               (int)((k_rem & 0xFFFFu) << 16),                     // tensor_dim0 lo
               (int)((k_rem >> 16) | ((rows_rem & 0xFFFFu) << 16)),// dim0 hi|dim1 lo
               (int)((rows_rem >> 16) | (32u << 16)),              // dim1 hi|tile_dim0=32
               (int)64,                                            // tile_dim1=64, tile_dim2=0
               (int)K,                                             // tensor_dim0_stride lo
               0, 0 };                                             // stride hi, dim1_stride
    v4i gz = {0, 0, 0, 0};
#if __clang_major__ >= 23
    v8i gz8 = {0, 0, 0, 0, 0, 0, 0, 0};
    __builtin_amdgcn_tensor_load_to_lds(g0, g1, gz, gz, gz8, 0);
#else
    __builtin_amdgcn_tensor_load_to_lds(g0, g1, gz, gz, 0);
#endif
}
#endif

// ---------------------------------------------------------------------------
// Generic WMMA GEMM: C[M,N] = A[M,K] @ W[K,N] + bias
//   A16   : A operand is f16 (staged via double-buffered TDM DMA if present)
//           else f32, converted while staging to LDS
//   RELU  : fuse ReLU
//   OUT16 : store f16 (else f32)
// Block 256 threads (8 waves). Tile 64x64, k-step 32.
// ---------------------------------------------------------------------------
template <int A16, int RELU, int OUT16>
__global__ void __launch_bounds__(256)
gemm_kernel(const void* __restrict__ Ap, const float* __restrict__ W,
            const float* __restrict__ bias, void* __restrict__ Cp,
            int M, int N, int K) {
    __shared__ __align__(16) _Float16 lds_a[2][64][40];   // [buf][m][k]
    __shared__ __align__(16) _Float16 lds_b[64][40];      // [n][k] (W transposed)

    const int tid  = threadIdx.x;
    const int lane = tid & 31;
    const int wave = tid >> 5;
    const int wr   = wave & 3;      // 4 row blocks of 16
    const int wc   = wave >> 2;     // 2 col blocks of 32
    const int bm   = blockIdx.y * 64;
    const int bn   = blockIdx.x * 64;

    const int r16 = lane & 15;
    const int hi  = lane >> 4;
    const int klo = hi * 8;

    // manual A staging pattern (fallback / f32 path)
    const int arow = tid >> 2;
    const int aseg = (tid & 3) * 8;
    int ar_g = bm + arow;
    if (ar_g >= M) ar_g = M - 1;

    // W staging: thread -> (k = tid&31, 8 consecutive n)
    const int wk = tid & 31;
    const int wn = (tid >> 5) * 8;

    v8f acc0 = {}; v8f acc1 = {};
    const int nk = K / 32;

#if HAVE_TDM
    if (A16 && wave == 0) {
        const _Float16* A = (const _Float16*)Ap;
        tdm_load_a_tile(&A[(size_t)bm * K], &lds_a[0][0][0],
                        (unsigned)(M - bm), (unsigned)K, (unsigned)K);
    }
#endif

    for (int i = 0; i < nk; ++i) {
        const int k0 = i * 32;
        const int cur = i & 1;

        if (A16) {
#if HAVE_TDM
            if (wave == 0) {
                const _Float16* A = (const _Float16*)Ap;
                if (i + 1 < nk) {
                    // prefetch next k-tile into the other buffer via TDM
                    tdm_load_a_tile(&A[(size_t)bm * K + (k0 + 32)],
                                    &lds_a[cur ^ 1][0][0],
                                    (unsigned)(M - bm), (unsigned)(K - (k0 + 32)),
                                    (unsigned)K);
                    __builtin_amdgcn_s_wait_tensorcnt(1);  // tile i resident
                } else {
                    __builtin_amdgcn_s_wait_tensorcnt(0);
                }
            }
#else
            const _Float16* A = (const _Float16*)Ap;
            *(v8h*)&lds_a[cur][arow][aseg] =
                *(const v8h*)&A[(size_t)ar_g * K + k0 + aseg];
#endif
        } else {
            const float* A = (const float*)Ap;
            const float* s = &A[(size_t)ar_g * K + k0 + aseg];
            v8h t;
#pragma unroll
            for (int j = 0; j < 8; ++j) t[j] = (_Float16)s[j];
            *(v8h*)&lds_a[cur][arow][aseg] = t;
            if (k0 + 32 < K)
                __builtin_prefetch(&A[(size_t)ar_g * K + k0 + 32 + aseg], 0, 1);
        }
        {
            const float* s = &W[(size_t)(k0 + wk) * N + bn + wn];
#pragma unroll
            for (int j = 0; j < 8; ++j) lds_b[wn + j][wk] = (_Float16)s[j];
        }
        __syncthreads();

        union { v16h v; v8h h[2]; } af, bf0, bf1;
        af.h[0]  = *(const v8h*)&lds_a[cur][wr * 16 + r16][klo];
        af.h[1]  = *(const v8h*)&lds_a[cur][wr * 16 + r16][klo + 16];
        bf0.h[0] = *(const v8h*)&lds_b[wc * 32 + r16][klo];
        bf0.h[1] = *(const v8h*)&lds_b[wc * 32 + r16][klo + 16];
        bf1.h[0] = *(const v8h*)&lds_b[wc * 32 + 16 + r16][klo];
        bf1.h[1] = *(const v8h*)&lds_b[wc * 32 + 16 + r16][klo + 16];

        acc0 = __builtin_amdgcn_wmma_f32_16x16x32_f16(false, af.v, false, bf0.v,
                                                      (short)0, acc0, false, false);
        acc1 = __builtin_amdgcn_wmma_f32_16x16x32_f16(false, af.v, false, bf1.v,
                                                      (short)0, acc1, false, false);
        __syncthreads();
    }

    // Epilogue: C layout -> VGPR g holds row (g + hi*8), col = lane&15
#pragma unroll
    for (int g = 0; g < 8; ++g) {
        int m = bm + wr * 16 + g + hi * 8;
        if (m >= M) continue;
        int n0 = bn + wc * 32 + r16;
        float c0 = acc0[g] + bias[n0];
        float c1 = acc1[g] + bias[n0 + 16];
        if (RELU) { c0 = fmaxf(c0, 0.f); c1 = fmaxf(c1, 0.f); }
        if (OUT16) {
            _Float16* Co = (_Float16*)Cp;
            Co[(size_t)m * N + n0]      = (_Float16)c0;
            Co[(size_t)m * N + n0 + 16] = (_Float16)c1;
        } else {
            float* Co = (float*)Cp;
            Co[(size_t)m * N + n0]      = c0;
            Co[(size_t)m * N + n0 + 16] = c1;
        }
    }
}

// ---------------------------------------------------------------------------
// V transpose:  vh (B,LQ,NH*DH) f16 -> vT (B,NH,DH,LQP) f16, zero padded
// ---------------------------------------------------------------------------
__global__ void vtrans_kernel(const _Float16* __restrict__ vh,
                              _Float16* __restrict__ vT) {
    int i = blockIdx.x * blockDim.x + threadIdx.x;  // B*NH*DH*LQP
    int q = i % LQP; int r = i / LQP;
    int dh = r % DHEAD; r /= DHEAD;
    int h = r % NHEAD; int b = r / NHEAD;
    _Float16 v = (_Float16)0.f;
    if (q < LQRY)
        v = vh[((size_t)(b * LQRY + q)) * D_MODEL + h * DHEAD + dh];
    vT[i] = v;
}

// ---------------------------------------------------------------------------
// Flash-style attention. Block = 128 (4 waves); wave -> one 16-query tile of
// one (b,h). Key tiles of 32: S via 2 WMMAs, online softmax, P through LDS
// (C-layout -> A-layout), O += P@V via 2 WMMAs against transposed V.
// ---------------------------------------------------------------------------
__global__ void __launch_bounds__(128)
attn_kernel(const _Float16* __restrict__ qh, const _Float16* __restrict__ kh,
            const _Float16* __restrict__ vT, _Float16* __restrict__ attn) {
    __shared__ __align__(16) _Float16 lds_p[4][16][40];

    const int wave = threadIdx.x >> 5;
    const int lane = threadIdx.x & 31;
    const int bh = blockIdx.x;             // b*NHEAD + h
    const int b = bh >> 3, h = bh & 7;
    const int nqt = (LQRY + 15) / 16;      // 57
    const int qt = blockIdx.y * 4 + wave;
    if (qt >= nqt) return;
    const int q0 = qt * 16;

    const int r16 = lane & 15, hi = lane >> 4, klo = hi * 8;

    union HF { v16h v; v8h h[2]; };
    HF qf;
    {
        int qr = q0 + r16; if (qr >= LQRY) qr = LQRY - 1;
        const _Float16* p = &qh[((size_t)(b * LQRY + qr)) * D_MODEL + h * DHEAD];
        qf.h[0] = *(const v8h*)(p + klo);
        qf.h[1] = *(const v8h*)(p + klo + 16);
    }

    float mrun[8], lrun[8];
    v8f o0 = {}, o1 = {};
#pragma unroll
    for (int g = 0; g < 8; ++g) { mrun[g] = -1e30f; lrun[g] = 0.f; }

    const float scale = 0.17677669529663688f;  // 1/sqrt(32)
    const _Float16* vTb = &vT[(size_t)bh * DHEAD * LQP];

    for (int kt = 0; kt < NKT; ++kt) {
        const int kb = kt * 32;
        HF kf0, kf1;
        {
            int kr0 = kb + r16;      if (kr0 >= LQRY) kr0 = LQRY - 1;
            int kr1 = kb + 16 + r16; if (kr1 >= LQRY) kr1 = LQRY - 1;
            const _Float16* p0 = &kh[((size_t)(b * LQRY + kr0)) * D_MODEL + h * DHEAD];
            const _Float16* p1 = &kh[((size_t)(b * LQRY + kr1)) * D_MODEL + h * DHEAD];
            kf0.h[0] = *(const v8h*)(p0 + klo); kf0.h[1] = *(const v8h*)(p0 + klo + 16);
            kf1.h[0] = *(const v8h*)(p1 + klo); kf1.h[1] = *(const v8h*)(p1 + klo + 16);
        }
        v8f zero = {};
        v8f s0 = __builtin_amdgcn_wmma_f32_16x16x32_f16(false, qf.v, false, kf0.v,
                                                        (short)0, zero, false, false);
        v8f s1 = __builtin_amdgcn_wmma_f32_16x16x32_f16(false, qf.v, false, kf1.v,
                                                        (short)0, zero, false, false);
        const bool ok0 = (kb + r16) < LQRY;
        const bool ok1 = (kb + 16 + r16) < LQRY;
#pragma unroll
        for (int g = 0; g < 8; ++g) {
            s0[g] = ok0 ? s0[g] * scale : -1e30f;
            s1[g] = ok1 ? s1[g] * scale : -1e30f;
        }
        // online softmax per row g (row lives across 16 lanes of one half-wave)
#pragma unroll
        for (int g = 0; g < 8; ++g) {
            float t = fmaxf(s0[g], s1[g]);
#pragma unroll
            for (int off = 8; off >= 1; off >>= 1)
                t = fmaxf(t, __shfl_xor(t, off, 16));
            float mnew = fmaxf(mrun[g], t);
            float p0 = __expf(s0[g] - mnew);
            float p1 = __expf(s1[g] - mnew);
            float rs = p0 + p1;
#pragma unroll
            for (int off = 8; off >= 1; off >>= 1)
                rs += __shfl_xor(rs, off, 16);
            float alpha = __expf(mrun[g] - mnew);
            lrun[g] = lrun[g] * alpha + rs;
            mrun[g] = mnew;
            o0[g] *= alpha; o1[g] *= alpha;
            int m = g + hi * 8;
            lds_p[wave][m][r16]      = (_Float16)p0;
            lds_p[wave][m][r16 + 16] = (_Float16)p1;
        }
        asm volatile("s_wait_dscnt 0" ::: "memory");

        HF pf, vf0, vf1;
        pf.h[0] = *(const v8h*)&lds_p[wave][r16][klo];
        pf.h[1] = *(const v8h*)&lds_p[wave][r16][klo + 16];
        const _Float16* v0 = &vTb[(size_t)r16 * LQP + kb];          // dh 0..15
        const _Float16* v1 = &vTb[(size_t)(16 + r16) * LQP + kb];   // dh 16..31
        vf0.h[0] = *(const v8h*)(v0 + klo); vf0.h[1] = *(const v8h*)(v0 + 16 + klo);
        vf1.h[0] = *(const v8h*)(v1 + klo); vf1.h[1] = *(const v8h*)(v1 + 16 + klo);

        o0 = __builtin_amdgcn_wmma_f32_16x16x32_f16(false, pf.v, false, vf0.v,
                                                    (short)0, o0, false, false);
        o1 = __builtin_amdgcn_wmma_f32_16x16x32_f16(false, pf.v, false, vf1.v,
                                                    (short)0, o1, false, false);
    }

#pragma unroll
    for (int g = 0; g < 8; ++g) {
        int qr = q0 + g + hi * 8;
        if (qr >= LQRY) continue;
        float inv = 1.f / lrun[g];
        size_t base = ((size_t)(b * LQRY + qr)) * D_MODEL + h * DHEAD;
        attn[base + r16]      = (_Float16)(o0[g] * inv);
        attn[base + 16 + r16] = (_Float16)(o1[g] * inv);
    }
}

// ---------------------------------------------------------------------------
// LayerNorm over D=256, one wave per row, fused residual add.
// Optional: f32 out, f16 out, f16(y + pos) out.
// ---------------------------------------------------------------------------
__global__ void __launch_bounds__(256)
ln_kernel(const float* __restrict__ a, const float* __restrict__ t2,
          const float* __restrict__ gamma, const float* __restrict__ beta,
          float* __restrict__ yf, _Float16* __restrict__ yh,
          const float* __restrict__ pos, _Float16* __restrict__ yph, int rows) {
    int wave = threadIdx.x >> 5, lane = threadIdx.x & 31;
    int r = blockIdx.x * 8 + wave;
    if (r >= rows) return;
    const float* pa = &a[(size_t)r * D_MODEL];
    const float* pb = &t2[(size_t)r * D_MODEL];
    float x[8]; float s = 0.f;
#pragma unroll
    for (int j = 0; j < 8; ++j) { x[j] = pa[lane * 8 + j] + pb[lane * 8 + j]; s += x[j]; }
#pragma unroll
    for (int off = 16; off >= 1; off >>= 1) s += __shfl_xor(s, off, 32);
    float mean = s * (1.f / 256.f);
    float v = 0.f;
#pragma unroll
    for (int j = 0; j < 8; ++j) { float d = x[j] - mean; v += d * d; }
#pragma unroll
    for (int off = 16; off >= 1; off >>= 1) v += __shfl_xor(v, off, 32);
    float rstd = rsqrtf(v * (1.f / 256.f) + 1e-5f);
#pragma unroll
    for (int j = 0; j < 8; ++j) {
        int c = lane * 8 + j;
        float y = (x[j] - mean) * rstd * gamma[c] + beta[c];
        if (yf)  yf[(size_t)r * D_MODEL + c] = y;
        if (yh)  yh[(size_t)r * D_MODEL + c] = (_Float16)y;
        if (yph) yph[(size_t)r * D_MODEL + c] =
                     (_Float16)(y + pos[(size_t)r * D_MODEL + c]);
    }
}

// ---------------------------------------------------------------------------
// MSDeformAttn sampling: one wave per (b,q,h), lane = channel dh.
// ---------------------------------------------------------------------------
__device__ __forceinline__ float samp1(const _Float16* vb, int xi, int yi,
                                       int W_, int H_) {
    bool ok = (xi >= 0) && (xi < W_) && (yi >= 0) && (yi < H_);
    int xc = min(max(xi, 0), W_ - 1);
    int yc = min(max(yi, 0), H_ - 1);
    float v = (float)vb[((size_t)(yc * W_ + xc)) * D_MODEL];
    return ok ? v : 0.f;
}

__global__ void __launch_bounds__(256)
msda_kernel(const float* __restrict__ off, const float* __restrict__ awl,
            const float* __restrict__ ref, const _Float16* __restrict__ value,
            _Float16* __restrict__ samp) {
    const int Hs[4] = {100, 50, 25, 13};
    const int Ws[4] = {100, 50, 25, 13};
    const int starts[4] = {0, 10000, 12500, 13125};
    int wave = threadIdx.x >> 5, lane = threadIdx.x & 31;
    int idx = blockIdx.x * 8 + wave;            // (b*LQ+q)*NH + h
    if (idx >= BATCH * LQRY * NHEAD) return;
    int h = idx & 7; int bq = idx >> 3;
    int b = bq / LQRY;

    float w[16]; float mx = -1e30f;
#pragma unroll
    for (int i = 0; i < 16; ++i) {
        w[i] = awl[(size_t)bq * 128 + h * 16 + i];
        mx = fmaxf(mx, w[i]);
    }
    float sum = 0.f;
#pragma unroll
    for (int i = 0; i < 16; ++i) { w[i] = __expf(w[i] - mx); sum += w[i]; }
    float inv = 1.f / sum;

    float acc = 0.f;
#pragma unroll
    for (int l = 0; l < NLVL; ++l) {
        const int H_ = Hs[l], W_ = Ws[l], st = starts[l];
        float rx = ref[((size_t)bq * NLVL + l) * 2 + 0];
        float ry = ref[((size_t)bq * NLVL + l) * 2 + 1];
        const _Float16* vb =
            &value[((size_t)(b * LINV + st)) * D_MODEL + h * DHEAD + lane];
#pragma unroll
        for (int p = 0; p < NPT; ++p) {
            int oi = ((h * NLVL + l) * NPT + p) * 2;
            float ox = off[(size_t)bq * 256 + oi];
            float oy = off[(size_t)bq * 256 + oi + 1];
            float px = (rx + ox / (float)W_) * W_ - 0.5f;
            float py = (ry + oy / (float)H_) * H_ - 0.5f;
            float fx = floorf(px), fy = floorf(py);
            float wx = px - fx, wy = py - fy;
            int x0 = (int)fx, y0 = (int)fy;
            float s = samp1(vb, x0,     y0,     W_, H_) * ((1.f - wx) * (1.f - wy))
                    + samp1(vb, x0 + 1, y0,     W_, H_) * (wx * (1.f - wy))
                    + samp1(vb, x0,     y0 + 1, W_, H_) * ((1.f - wx) * wy)
                    + samp1(vb, x0 + 1, y0 + 1, W_, H_) * (wx * wy);
            acc += s * w[l * NPT + p] * inv;
        }
    }
    samp[(size_t)bq * D_MODEL + h * DHEAD + lane] = (_Float16)acc;
}

// ---------------------------------------------------------------------------
// Host orchestration
// ---------------------------------------------------------------------------
extern "C" void kernel_launch(void* const* d_in, const int* in_sizes, int n_in,
                              void* d_out, int out_size, void* d_ws, size_t ws_size,
                              hipStream_t stream) {
    const float* tgt   = (const float*)d_in[0];
    const float* qpos  = (const float*)d_in[1];
    const float* ref   = (const float*)d_in[2];
    const float* src   = (const float*)d_in[3];
    const float* Wq    = (const float*)d_in[6];
    const float* bq    = (const float*)d_in[7];
    const float* Wk    = (const float*)d_in[8];
    const float* bk    = (const float*)d_in[9];
    const float* Wv    = (const float*)d_in[10];
    const float* bv    = (const float*)d_in[11];
    const float* Wo    = (const float*)d_in[12];
    const float* bo    = (const float*)d_in[13];
    const float* off_W = (const float*)d_in[14];
    const float* off_b = (const float*)d_in[15];
    const float* aw_W  = (const float*)d_in[16];
    const float* aw_b  = (const float*)d_in[17];
    const float* val_W = (const float*)d_in[18];
    const float* val_b = (const float*)d_in[19];
    const float* out_W = (const float*)d_in[20];
    const float* out_b = (const float*)d_in[21];
    const float* g1    = (const float*)d_in[22];
    const float* bn1   = (const float*)d_in[23];
    const float* g2    = (const float*)d_in[24];
    const float* bn2   = (const float*)d_in[25];
    const float* g3    = (const float*)d_in[26];
    const float* bn3   = (const float*)d_in[27];
    const float* fW1   = (const float*)d_in[28];
    const float* fb1   = (const float*)d_in[29];
    const float* fW2   = (const float*)d_in[30];
    const float* fb2   = (const float*)d_in[31];

    char* ws = (char*)d_ws;
    size_t o = 0;
    auto alloc = [&](size_t bytes) -> void* {
        void* p = ws + o;
        o = (o + bytes + 255) & ~(size_t)255;
        return p;
    };

    _Float16* q1h    = (_Float16*)alloc((size_t)NTOK * 256 * 2);
    _Float16* tgth   = (_Float16*)alloc((size_t)NTOK * 256 * 2);
    _Float16* qh_    = (_Float16*)alloc((size_t)NTOK * 256 * 2);
    _Float16* kh_    = (_Float16*)alloc((size_t)NTOK * 256 * 2);
    _Float16* vh_    = (_Float16*)alloc((size_t)NTOK * 256 * 2);
    _Float16* valueh = (_Float16*)alloc((size_t)NVAL * 256 * 2);
    _Float16* vT     = (_Float16*)alloc((size_t)64 * DHEAD * LQP * 2);
    float*    t2     = (float*)   alloc((size_t)NTOK * 256 * 4);
    float*    x1     = (float*)   alloc((size_t)NTOK * 256 * 4);
    _Float16* qposh  = (_Float16*)alloc((size_t)NTOK * 256 * 2);
    float*    offb   = (float*)   alloc((size_t)NTOK * 256 * 4);
    float*    awb    = (float*)   alloc((size_t)NTOK * 128 * 4);
    _Float16* ffnh   = (_Float16*)alloc((size_t)NTOK * DFF * 2);
    // aliases (lifetimes do not overlap):
    _Float16* attnh = tgth;   // tgth last read by Wv GEMM
    _Float16* samph = qh_;    // qh_ last read by attention
    _Float16* x2h   = q1h;    // q1h last read by Wq/Wk GEMMs
    float*    x2    = x1;     // LN2 in-place over x1 (elementwise-safe)

    const int gElem = (NTOK * 256 + 255) / 256;
    const dim3 blk256(256), blk128(128);
    const int gM   = (NTOK + 63) / 64;   // 113
    const int gMv  = (NVAL + 63) / 64;   // 1662

    // 1) stage f16 operands
    addcvt_kernel<<<gElem, blk256, 0, stream>>>(tgt, qpos, q1h, NTOK * 256);
    addcvt_kernel<<<gElem, blk256, 0, stream>>>(tgt, nullptr, tgth, NTOK * 256);

    // 2) self-attention projections (f16 out)
    gemm_kernel<1, 0, 1><<<dim3(4, gM), blk256, 0, stream>>>(q1h,  Wq, bq, qh_, NTOK, 256, 256);
    gemm_kernel<1, 0, 1><<<dim3(4, gM), blk256, 0, stream>>>(q1h,  Wk, bk, kh_, NTOK, 256, 256);
    gemm_kernel<1, 0, 1><<<dim3(4, gM), blk256, 0, stream>>>(tgth, Wv, bv, vh_, NTOK, 256, 256);

    // 3) deform-attn value projection (A is f32 src, converted on the fly)
    gemm_kernel<0, 0, 1><<<dim3(4, gMv), blk256, 0, stream>>>(src, val_W, val_b, valueh,
                                                              NVAL, 256, 256);

    // 4) transpose V for attention B-fragments
    vtrans_kernel<<<(64 * DHEAD * LQP) / 256, blk256, 0, stream>>>(vh_, vT);

    // 5) flash attention
    attn_kernel<<<dim3(64, 15), blk128, 0, stream>>>(qh_, kh_, vT, attnh);

    // 6) attention output projection + LN1 (also emits f16(x1+qpos))
    gemm_kernel<1, 0, 0><<<dim3(4, gM), blk256, 0, stream>>>(attnh, Wo, bo, t2, NTOK, 256, 256);
    ln_kernel<<<(NTOK + 7) / 8, blk256, 0, stream>>>(tgt, t2, g1, bn1,
                                                     x1, nullptr, qpos, qposh, NTOK);

    // 7) sampling offsets / attention weights
    gemm_kernel<1, 0, 0><<<dim3(4, gM), blk256, 0, stream>>>(qposh, off_W, off_b, offb,
                                                             NTOK, 256, 256);
    gemm_kernel<1, 0, 0><<<dim3(2, gM), blk256, 0, stream>>>(qposh, aw_W, aw_b, awb,
                                                             NTOK, 128, 256);

    // 8) deformable sampling
    msda_kernel<<<(BATCH * LQRY * NHEAD) / 8, blk256, 0, stream>>>(offb, awb, ref,
                                                                   valueh, samph);

    // 9) deform-attn output projection + LN2
    gemm_kernel<1, 0, 0><<<dim3(4, gM), blk256, 0, stream>>>(samph, out_W, out_b, t2,
                                                             NTOK, 256, 256);
    ln_kernel<<<(NTOK + 7) / 8, blk256, 0, stream>>>(x1, t2, g2, bn2,
                                                     x2, x2h, nullptr, nullptr, NTOK);

    // 10) FFN (relu fused into first GEMM, f16 hidden)
    gemm_kernel<1, 1, 1><<<dim3(16, gM), blk256, 0, stream>>>(x2h, fW1, fb1, ffnh,
                                                              NTOK, DFF, 256);
    gemm_kernel<1, 0, 0><<<dim3(4, gM), blk256, 0, stream>>>(ffnh, fW2, fb2, t2,
                                                             NTOK, 256, DFF);

    // 11) LN3 -> final output (f32)
    ln_kernel<<<(NTOK + 7) / 8, blk256, 0, stream>>>(x2, t2, g3, bn3,
                                                     (float*)d_out, nullptr, nullptr,
                                                     nullptr, NTOK);
}
